// G_E_GNN_83674552860993
// MI455X (gfx1250) — compile-verified, matrix operand order checked by reading, not run on hardware
//
#include <hip/hip_runtime.h>

// ---------------------------------------------------------------------------
// MI455X (gfx1250) implementation of the chain-graph GNN step.
//   * all 128-wide GEMMs via v_wmma_f32_16x16x32_bf16 (f32 accumulate)
//   * weights repacked once per call into WMMA B-fragment order, staged in
//     dynamic LDS (~302KB for the edge kernel -- CDNA5 has 320KB/WGP)
//   * wave32: one 16-row node tile per wave; C/D->A transposes stream through
//     per-wave LDS buffers (same-wave DS ordering)
//   * B fragments software-pipelined (rotating double buffer) so ds_load
//     latency overlaps the WMMA stream instead of s_wait_dscnt 0 per WMMA
//   * register-pressure-shaped: peak accumulators kept < 256 VGPRs
//   * BatchNorm via partial-sum + reduce kernels (no atomics, deterministic)
// Workspace use: ~135.4 MB (agg/x1 buffer reused in place).
// ---------------------------------------------------------------------------

#define NF   128
#define DSZ  4
#define TLEN 131072

typedef __bf16 bf16;
typedef __attribute__((ext_vector_type(16))) __bf16 v16bf;
typedef __attribute__((ext_vector_type(8)))  float  v8f;
typedef __attribute__((ext_vector_type(4)))  float  f32x4;
typedef __attribute__((ext_vector_type(4)))  unsigned int u32x4;

union FragU { v16bf v; u32x4 q[2]; };

static __device__ __forceinline__ v8f wmma_bf16(v16bf a, v16bf b, v8f c) {
  // (neg_a, A, neg_b, B, c_mod, C, reuse_a, reuse_b)
  return __builtin_amdgcn_wmma_f32_16x16x32_bf16(false, a, false, b, (short)0, c, false, false);
}
// 32 contiguous bytes (fragment-packed storage): two b128 loads
static __device__ __forceinline__ v16bf load_frag32(const bf16* p) {
  FragU u; u.q[0] = *(const u32x4*)p; u.q[1] = *(const u32x4*)(p + 8); return u.v;
}
// A-row fragment: halves {0..7} at p, {16..23} at p+16 (per 16-bit A layout)
static __device__ __forceinline__ v16bf load_arow(const bf16* p) {
  FragU u; u.q[0] = *(const u32x4*)p; u.q[1] = *(const u32x4*)(p + 16); return u.v;
}
static __device__ __forceinline__ float leaky(float x) { return x >= 0.f ? x : 0.01f * x; }
static __device__ __forceinline__ float sigmf(float x) { return 1.f / (1.f + __expf(-x)); }

// ---------------------------------------------------------------------------
// Weight packing: B[k][n] = W[n][k] into per-fragment contiguous order.
// Fragment (kc,nt): lane<16 -> N=nt*16+lane, K = kc*32 + {0..7,16..23}
//                   lane>=16 -> same N set, K offset +8 (mirrors A layout).
// ---------------------------------------------------------------------------
__global__ void pack_frags(const float* __restrict__ W, bf16* __restrict__ dst,
                           int Kdim, int nKC, int nNT) {
  int total = nKC * nNT * 512;
  for (int id = blockIdx.x * blockDim.x + threadIdx.x; id < total;
       id += gridDim.x * blockDim.x) {
    int hh = id & 15, lane = (id >> 4) & 31;
    int fragIdx = id >> 9;
    int nt = fragIdx % nNT, kc = fragIdx / nNT;
    int N = nt * 16 + (lane & 15);
    int K = kc * 32 + (hh < 8 ? hh : hh + 8) + (lane < 16 ? 0 : 8);
    dst[id] = (bf16)W[(size_t)N * Kdim + K];
  }
}

// attr-tail weights: dst[nt][i][j] = W1[nt*16+i][256+j] (f32, K=4 VALU tail)
__global__ void pack_attrw(const float* __restrict__ W1, float* __restrict__ dst) {
  int id = blockIdx.x * blockDim.x + threadIdx.x;
  if (id < 512) {
    int j = id & 3, i = (id >> 2) & 15, nt = id >> 6;
    dst[id] = W1[(size_t)(nt * 16 + i) * 260 + 256 + j];
  }
}

// hy[t][n] = fc_init_w[n] . inv[t] + b[n]   (inv = replicate-padded diffs)
__global__ void prep_hy(const float* __restrict__ meas, const float* __restrict__ w,
                        const float* __restrict__ b, bf16* __restrict__ hy) {
  __shared__ float sw[NF * 4 + NF];
  for (int i = threadIdx.x; i < NF * 4 + NF; i += blockDim.x)
    sw[i] = (i < NF * 4) ? w[i] : b[i - NF * 4];
  __syncthreads();
  int t = blockIdx.x * blockDim.x + threadIdx.x;
  if (t >= TLEN) return;
  int tm = t > 0 ? t - 1 : 0, tp = t < TLEN - 1 ? t + 1 : TLEN - 1;
  float m0 = meas[t], m1 = meas[TLEN + t];
  float i0 = m0 - meas[tm], i1 = m1 - meas[TLEN + tm];
  float i2 = meas[tp] - m0, i3 = meas[TLEN + tp] - m1;
  bf16* o = hy + (size_t)t * NF;
  for (int n = 0; n < NF; ++n) {
    float v = sw[n*4]*i0 + sw[n*4+1]*i1 + sw[n*4+2]*i2 + sw[n*4+3]*i3 + sw[NF*4 + n];
    o[n] = (bf16)v;
  }
}

// h [NF,T] f32  ->  hT [T,NF] bf16 (row-contiguous A-operand source)
__global__ void prep_ht(const float* __restrict__ h, bf16* __restrict__ hT) {
  __shared__ float tile[32][33];
  int f0 = blockIdx.y * 32, t0 = blockIdx.x * 32;
  int lx = threadIdx.x, ly = threadIdx.y;
  for (int i = 0; i < 32; i += 8)
    tile[ly + i][lx] = h[(size_t)(f0 + ly + i) * TLEN + t0 + lx];
  __syncthreads();
  for (int i = 0; i < 32; i += 8)
    hT[(size_t)(t0 + ly + i) * NF + f0 + lx] = (bf16)tile[lx][ly + i];
}

// column partial sums / sumsq for BatchNorm (512 blocks x 128 cols)
__global__ void col_stats(const float* __restrict__ X, float* __restrict__ part) {
  int c = threadIdx.x;
  int nr = TLEN / gridDim.x;
  int r0 = blockIdx.x * nr;
  float s = 0.f, s2 = 0.f;
  for (int r = 0; r < nr; ++r) {
    float v = X[(size_t)(r0 + r) * NF + c];
    s += v; s2 += v * v;
  }
  part[blockIdx.x * 256 + c] = s;
  part[blockIdx.x * 256 + 128 + c] = s2;
}

// fold mean/var/gamma/beta into per-column scale+shift
__global__ void stats_reduce(const float* __restrict__ part, int nblk,
                             const float* __restrict__ gamma, const float* __restrict__ beta,
                             float* __restrict__ st) {
  int c = threadIdx.x;
  float s = 0.f, s2 = 0.f;
  for (int b = 0; b < nblk; ++b) { s += part[b * 256 + c]; s2 += part[b * 256 + 128 + c]; }
  float m = s / (float)TLEN;
  float v = s2 / (float)TLEN - m * m;
  float sc = gamma[c] * rsqrtf(v + 1e-5f);
  st[c] = sc;
  st[128 + c] = beta[c] - m * sc;
}

// ---------------------------------------------------------------------------
// Edge MLP: [self(128) | nbr(128) | attr(4)] -> 128 (leaky) -> 128 (leaky),
// masked at chain boundary, accumulated into S.
// Layer-1 runs as one software-pipelined 32-WMMA stream per half (self+nbr
// flattened); each half immediately feeds its two layer-2 K-chunks.
// ---------------------------------------------------------------------------
static __device__ __forceinline__ void edge_mlp(
    v8f (&S)[8],
    const bf16* selfRow, const bf16* nbrRow,
    const float* __restrict__ ea,
    const bf16* sL1, const float* sAW, const bf16* sL2,
    const float* __restrict__ b1, const float* __restrict__ b2,
    bf16* xb, int t0, int li, int koff, int lane, int killRR, int killHalf) {
  // attr values (K 256..259), shared by both halves
  float at[4][8];
#pragma unroll
  for (int j = 0; j < 4; ++j) {
    const float* p = ea + (size_t)j * TLEN + t0 + koff;
    f32x4 q0 = *(const f32x4*)p, q1 = *(const f32x4*)(p + 4);
#pragma unroll
    for (int rr = 0; rr < 4; ++rr) { at[j][rr] = q0[rr]; at[j][rr + 4] = q1[rr]; }
  }
  v8f A2[8] = {};
#pragma unroll
  for (int half = 0; half < 2; ++half) {
    v8f A1[4] = {};
    {
      // flattened self(kc 0..3) + neighbor(kc 4..7) stream, double-buffered
      auto baddr = [&](int i) {
        int kc = i >> 2, q = i & 3;
        return sL1 + ((size_t)(kc * 8 + half * 4 + q) * 32 + lane) * 16;
      };
      v16bf a = load_arow(selfRow + koff);
      v16bf b = load_frag32(baddr(0));
#pragma unroll
      for (int i = 0; i < 32; ++i) {
        v16bf bn = (i < 31) ? load_frag32(baddr(i + 1)) : b;
        v16bf an = a;
        if ((i & 3) == 3 && i < 31) {
          int kcn = (i + 1) >> 2;
          an = load_arow(((kcn < 4) ? (selfRow + kcn * 32) : (nbrRow + (kcn - 4) * 32)) + koff);
        }
        A1[i & 3] = wmma_bf16(a, b, A1[i & 3]);
        a = an; b = bn;
      }
    }
    // attr tail (VALU) + bias + leaky
#pragma unroll
    for (int q = 0; q < 4; ++q) {
      int nt = half * 4 + q;
      f32x4 w4 = *(const f32x4*)(sAW + nt * 64 + li * 4);
      float bb = b1[nt * 16 + li];
#pragma unroll
      for (int rr = 0; rr < 8; ++rr) {
        float v = A1[q][rr] + at[0][rr] * w4[0] + at[1][rr] * w4[1]
                            + at[2][rr] * w4[2] + at[3][rr] * w4[3] + bb;
        A1[q][rr] = leaky(v);
      }
    }
    // layer 2 for this half's two K-chunks (C/D->A via per-wave LDS buffer)
#pragma unroll
    for (int p = 0; p < 2; ++p) {
      int kc2 = half * 2 + p;
#pragma unroll
      for (int q = 0; q < 2; ++q) {
        int src = p * 2 + q;
#pragma unroll
        for (int rr = 0; rr < 8; ++rr)
          xb[(rr + koff) * 32 + q * 16 + li] = (bf16)A1[src][rr];
      }
      v16bf a = load_arow(xb + li * 32 + koff);
      v16bf b = load_frag32(sL2 + ((size_t)(kc2 * 8) * 32 + lane) * 16);
#pragma unroll
      for (int nt = 0; nt < 8; ++nt) {
        v16bf bn = (nt < 7)
            ? load_frag32(sL2 + ((size_t)(kc2 * 8 + nt + 1) * 32 + lane) * 16) : b;
        A2[nt] = wmma_bf16(a, b, A2[nt]);
        b = bn;
      }
    }
  }
#pragma unroll
  for (int nt = 0; nt < 8; ++nt) {
    float bb = b2[nt * 16 + li];
#pragma unroll
    for (int rr = 0; rr < 8; ++rr) {
      float x = leaky(A2[nt][rr] + bb);
      if (rr == killRR && (lane >> 4) == killHalf) x = 0.f;   // boundary edge
      S[nt][rr] += x;
    }
  }
}

// ---------------------------------------------------------------------------
// Kernel 1: edges + segment-sum + node linear -> buf [T,NF] (pre-BN)
// Dynamic LDS image (copied from ws): 3xL1 frags | 3xL2 frags | 3xattrW,
// then 1KB/wave transpose buffers.  Total 309248 B.
// ---------------------------------------------------------------------------
__global__ __launch_bounds__(256) void gnn_edge_node(
    const bf16* __restrict__ hT, const bf16* __restrict__ hy,
    const float* __restrict__ ea_r, const float* __restrict__ ea_u, const float* __restrict__ ea_l,
    const float* __restrict__ fragsSrc, const bf16* __restrict__ nodeF,
    const float* __restrict__ b1u, const float* __restrict__ b2u,
    const float* __restrict__ b1r, const float* __restrict__ b2r,
    const float* __restrict__ b1l, const float* __restrict__ b2l,
    const float* __restrict__ node_b, float* __restrict__ buf) {
  extern __shared__ char smem[];
  bf16*  sFrag  = (bf16*)smem;                 // 147456 halves (L1x3 | L2x3)
  float* sAttrW = (float*)(smem + 294912);     // 1536 f32
  bf16*  sXbAll = (bf16*)(smem + 301056);      // 512 halves per wave
  {
    const u32x4* src = (const u32x4*)fragsSrc;
    u32x4* dst = (u32x4*)smem;
    for (int i = threadIdx.x; i < 301056 / 16; i += blockDim.x) dst[i] = src[i];
  }
  __syncthreads();

  const int lane = threadIdx.x & 31, wid = threadIdx.x >> 5;
  const int li = lane & 15, koff = (lane >> 4) * 8;
  bf16* xb = sXbAll + wid * 512;
  const int nw = gridDim.x * (blockDim.x >> 5);
  const int lastTile = TLEN / 16 - 1;

  for (int tile = blockIdx.x * (blockDim.x >> 5) + wid; tile <= lastTile; tile += nw) {
    const int t0 = tile * 16, tR = t0 + li;
    if (tile + nw <= lastTile)
      __builtin_prefetch(hT + (size_t)(tile + nw) * 16 * NF, 0, 1);  // global_prefetch_b8
    v8f S[8] = {};
    // edge u: neighbor is measurement node hy[i]
    edge_mlp(S, hT + (size_t)tR * NF, hy + (size_t)tR * NF, ea_u,
             sFrag + 0, sAttrW + 0, sFrag + 98304,
             b1u, b2u, xb, t0, li, koff, lane, -1, 0);
    // edge r: from i-1, invalid at i==0
    edge_mlp(S, hT + (size_t)tR * NF, hT + (size_t)(tR > 0 ? tR - 1 : 0) * NF, ea_r,
             sFrag + 32768, sAttrW + 512, sFrag + 114688,
             b1r, b2r, xb, t0, li, koff, lane, (t0 == 0) ? 0 : -1, 0);
    // edge l: from i+1, invalid at i==T-1
    edge_mlp(S, hT + (size_t)tR * NF, hT + (size_t)(tR < TLEN - 1 ? tR + 1 : TLEN - 1) * NF, ea_l,
             sFrag + 65536, sAttrW + 1024, sFrag + 131072,
             b1l, b2l, xb, t0, li, koff, lane, (tile == lastTile) ? 7 : -1, 1);
    // node linear (B frags from L2-resident ws), double-buffered
    v8f AN[8] = {};
#pragma unroll
    for (int kc = 0; kc < 4; ++kc) {
#pragma unroll
      for (int p = 0; p < 2; ++p) {
        int nt = kc * 2 + p;
#pragma unroll
        for (int rr = 0; rr < 8; ++rr) xb[(rr + koff) * 32 + p * 16 + li] = (bf16)S[nt][rr];
      }
      v16bf a = load_arow(xb + li * 32 + koff);
      v16bf b = load_frag32(nodeF + ((size_t)(kc * 8) * 32 + lane) * 16);
#pragma unroll
      for (int nt = 0; nt < 8; ++nt) {
        v16bf bn = (nt < 7)
            ? load_frag32(nodeF + ((size_t)(kc * 8 + nt + 1) * 32 + lane) * 16) : b;
        AN[nt] = wmma_bf16(a, b, AN[nt]);
        b = bn;
      }
    }
#pragma unroll
    for (int nt = 0; nt < 8; ++nt) {
      float bb = node_b[nt * 16 + li];
#pragma unroll
      for (int rr = 0; rr < 8; ++rr)
        buf[(size_t)(t0 + rr + koff) * NF + nt * 16 + li] = AN[nt][rr] + bb;
    }
  }
}

// ---------------------------------------------------------------------------
// Kernel 3: BN+leaky -> GRU cell -> h_out [NF,T]; dec1 linear -> buf (x1)
// Column-chunked GRU: per 16-col chunk only r/z/n/hn accs live (32 VGPRs);
// gi accumulates on top of gh (free add for the r/z gate sums); the 24-WMMA
// gate stream is software-pipelined over its B fragments.
// hnew -> per-wave 16x128 bf16 LDS buffer -> A-frags for dec1 GEMM.
// LDS: Wih|Whh|dec1 frags (229376B) + bn scale/shift + 4KB/wave hbuf.
// ---------------------------------------------------------------------------
__global__ __launch_bounds__(256) void gnn_gru_dec1(
    float* buf, const bf16* __restrict__ hT, const float* __restrict__ h,
    const float* __restrict__ fragsSrc, const float* __restrict__ st,
    const float* __restrict__ b_ih, const float* __restrict__ b_hh,
    const float* __restrict__ d1b, float* __restrict__ outH) {
  extern __shared__ char smem[];
  bf16*  sWih = (bf16*)smem;            // 49152 halves
  bf16*  sWhh = sWih + 49152;           // 49152 halves
  bf16*  sD1  = sWih + 98304;           // 16384 halves
  float* sST  = (float*)(smem + 229376);
  bf16*  sHbAll = (bf16*)(smem + 230400);   // 2048 halves per wave
  {
    const u32x4* src = (const u32x4*)fragsSrc;
    u32x4* dst = (u32x4*)smem;
    for (int i = threadIdx.x; i < 229376 / 16; i += blockDim.x) dst[i] = src[i];
  }
  for (int i = threadIdx.x; i < 256; i += blockDim.x) sST[i] = st[i];
  __syncthreads();

  const int lane = threadIdx.x & 31, wid = threadIdx.x >> 5;
  const int li = lane & 15, koff = (lane >> 4) * 8;
  bf16* hb = sHbAll + wid * 2048;
  const int nw = gridDim.x * (blockDim.x >> 5);

  for (int tile = blockIdx.x * (blockDim.x >> 5) + wid; tile < TLEN / 16; tile += nw) {
    const int t0 = tile * 16, tR = t0 + li;
    v16bf aG[4], aH[4];
#pragma unroll
    for (int kc = 0; kc < 4; ++kc) {
      const float* p = buf + (size_t)tR * NF + kc * 32 + koff;
      f32x4 q0 = *(const f32x4*)p,        q1 = *(const f32x4*)(p + 4);
      f32x4 q2 = *(const f32x4*)(p + 16), q3 = *(const f32x4*)(p + 20);
      float x[16];
#pragma unroll
      for (int m = 0; m < 4; ++m) { x[m] = q0[m]; x[4+m] = q1[m]; x[8+m] = q2[m]; x[12+m] = q3[m]; }
      const float* ss = sST + kc * 32 + koff;
      FragU u;
#pragma unroll
      for (int m = 0; m < 16; ++m) {
        int fo = (m < 8) ? m : m + 8;
        u.v[m] = (bf16)leaky(x[m] * ss[fo] + ss[128 + fo]);   // BN + leaky
      }
      aG[kc] = u.v;
      aH[kc] = load_arow(hT + (size_t)tR * NF + kc * 32 + koff);
    }
    // GRU per 16-column chunk
#pragma unroll
    for (int c = 0; c < 8; ++c) {
      v8f aR = {}, aZ = {}, aN = {}, aHn = {};
      {
        // idx 0..23: kc = idx/6; slots {Whh:r,z,hn | Wih:r,z,n}
        auto gaddr = [&](int idx) {
          int kc = idx / 6, s = idx % 6;
          const bf16* base = (s < 3) ? sWhh : sWih;
          int nt = ((s == 0 || s == 3) ? c : (s == 1 || s == 4) ? 8 + c : 16 + c);
          return base + ((size_t)(kc * 24 + nt) * 32 + lane) * 16;
        };
        v16bf b = load_frag32(gaddr(0));
#pragma unroll
        for (int idx = 0; idx < 24; ++idx) {
          v16bf bn = (idx < 23) ? load_frag32(gaddr(idx + 1)) : b;
          int kc = idx / 6, s = idx % 6;
          v16bf av = (s < 3) ? aH[kc] : aG[kc];
          if (s == 0 || s == 3)      aR  = wmma_bf16(av, b, aR);
          else if (s == 1 || s == 4) aZ  = wmma_bf16(av, b, aZ);
          else if (s == 2)           aHn = wmma_bf16(av, b, aHn);
          else                       aN  = wmma_bf16(av, b, aN);
          b = bn;
        }
      }
      float br  = b_ih[c * 16 + li] + b_hh[c * 16 + li];
      float bz  = b_ih[128 + c * 16 + li] + b_hh[128 + c * 16 + li];
      float bn_ = b_ih[256 + c * 16 + li];
      float bh  = b_hh[256 + c * 16 + li];
      const float* hp = h + (size_t)(c * 16 + li) * TLEN + t0 + koff;
      f32x4 h0 = *(const f32x4*)hp, h1 = *(const f32x4*)(hp + 4);
      float hn[8];
#pragma unroll
      for (int rr = 0; rr < 8; ++rr) {
        float hold = (rr < 4) ? h0[rr] : h1[rr - 4];
        float r = sigmf(aR[rr] + br);
        float z = sigmf(aZ[rr] + bz);
        float n = tanhf(aN[rr] + bn_ + r * (aHn[rr] + bh));
        hn[rr] = (1.f - z) * n + z * hold;
        hb[(rr + koff) * NF + c * 16 + li] = (bf16)hn[rr];   // stash for dec1
      }
      float* op = outH + (size_t)(c * 16 + li) * TLEN + t0 + koff;
      f32x4 o0 = {hn[0], hn[1], hn[2], hn[3]};
      f32x4 o1 = {hn[4], hn[5], hn[6], hn[7]};
      *(f32x4*)op = o0; *(f32x4*)(op + 4) = o1;
    }
    // dec1 linear: x1 = hnew @ dec1_w.T + b  (A-frags straight from hbuf),
    // flattened 32-WMMA stream with double-buffered B frags
    v8f accD[8] = {};
    {
      v16bf a = load_arow(hb + li * NF + koff);
      v16bf b = load_frag32(sD1 + (size_t)lane * 16);
#pragma unroll
      for (int i = 0; i < 32; ++i) {
        int nt = i & 7;
        v16bf bn = (i < 31) ? load_frag32(sD1 + ((size_t)(i + 1) * 32 + lane) * 16) : b;
        v16bf an = a;
        if ((i & 7) == 7 && i < 31)
          an = load_arow(hb + li * NF + (((i + 1) >> 3) * 32) + koff);
        accD[nt] = wmma_bf16(a, b, accD[nt]);
        a = an; b = bn;
      }
    }
#pragma unroll
    for (int nt = 0; nt < 8; ++nt) {
      float bb = d1b[nt * 16 + li];
#pragma unroll
      for (int rr = 0; rr < 8; ++rr)
        buf[(size_t)(t0 + rr + koff) * NF + nt * 16 + li] = accD[nt][rr] + bb;
    }
  }
}

// dec2: BN+relu then 128->4 dot per row (N=4 too skinny for WMMA)
__global__ __launch_bounds__(256) void gnn_dec2(
    const float* __restrict__ x1, const float* __restrict__ st,
    const float* __restrict__ w2, const float* __restrict__ b2,
    float* __restrict__ dec) {
  __shared__ float sw[4 * NF + 256];
  for (int i = threadIdx.x; i < 4 * NF; i += blockDim.x) sw[i] = w2[i];
  for (int i = threadIdx.x; i < 256; i += blockDim.x) sw[4 * NF + i] = st[i];
  __syncthreads();
  int lane = threadIdx.x & 31, wid = threadIdx.x >> 5;
  size_t row = (size_t)blockIdx.x * (blockDim.x >> 5) + wid;
  if (row >= TLEN) return;
  f32x4 q = *(const f32x4*)(x1 + row * NF + lane * 4);
  float a[4] = {0.f, 0.f, 0.f, 0.f};
#pragma unroll
  for (int k = 0; k < 4; ++k) {
    int f = lane * 4 + k;
    float v = q[k] * sw[4 * NF + f] + sw[4 * NF + 128 + f];
    v = v > 0.f ? v : 0.f;
#pragma unroll
    for (int d = 0; d < 4; ++d) a[d] += v * sw[d * NF + f];
  }
#pragma unroll
  for (int d = 0; d < 4; ++d) {
    float s = a[d];
#pragma unroll
    for (int off = 16; off; off >>= 1) s += __shfl_down(s, off, 32);
    if (lane == 0) dec[(size_t)d * TLEN + row] = s + b2[d];
  }
}

// ---------------------------------------------------------------------------
extern "C" void kernel_launch(void* const* d_in, const int* in_sizes, int n_in,
                              void* d_out, int out_size, void* d_ws, size_t ws_size,
                              hipStream_t stream) {
  (void)in_sizes; (void)n_in; (void)out_size; (void)ws_size;
  const float* h    = (const float*)d_in[0];
  const float* meas = (const float*)d_in[1];
  const float* ea_r = (const float*)d_in[2];
  const float* ea_u = (const float*)d_in[3];
  const float* ea_l = (const float*)d_in[4];
  const float* fcw  = (const float*)d_in[5];
  const float* fcb  = (const float*)d_in[6];
  const float* Wl1 = (const float*)d_in[7],  *bl1 = (const float*)d_in[8];
  const float* Wl2 = (const float*)d_in[9],  *bl2 = (const float*)d_in[10];
  const float* Wr1 = (const float*)d_in[11], *br1 = (const float*)d_in[12];
  const float* Wr2 = (const float*)d_in[13], *br2 = (const float*)d_in[14];
  const float* Wu1 = (const float*)d_in[15], *bu1 = (const float*)d_in[16];
  const float* Wu2 = (const float*)d_in[17], *bu2 = (const float*)d_in[18];
  const float* node_w = (const float*)d_in[19], *node_b = (const float*)d_in[20];
  const float* ngam = (const float*)d_in[21], *nbet = (const float*)d_in[22];
  const float* W_ih = (const float*)d_in[23], *W_hh = (const float*)d_in[24];
  const float* b_ih = (const float*)d_in[25], *b_hh = (const float*)d_in[26];
  const float* d1w = (const float*)d_in[27], *d1b = (const float*)d_in[28];
  const float* dgam = (const float*)d_in[29], *dbet = (const float*)d_in[30];
  const float* d2w = (const float*)d_in[31], *d2b = (const float*)d_in[32];

  char* ws = (char*)d_ws;
  float* buf   = (float*)(ws + 0);                 // [T,NF] f32: agg_lin then x1
  bf16*  hyB   = (bf16*)(ws + 67108864);           // [T,NF] bf16
  bf16*  hTB   = (bf16*)(ws + 100663296);          // [T,NF] bf16
  char*  frag1 = ws + 134217728;                   // 301056 B (kernel-1 LDS image)
  bf16*  nodeF = (bf16*)(ws + 134518784);          // 32768 B
  char*  frag3 = ws + 134551552;                   // 229376 B (kernel-3 LDS image)
  float* part  = (float*)(ws + 134780928);         // 524288 B
  float* st1   = (float*)(ws + 135305216);
  float* st2   = (float*)(ws + 135306240);

  // weight packing (order must match the LDS images)
  pack_frags<<<128, 256, 0, stream>>>(Wu1, (bf16*)(frag1 + 0),      260, 8, 8);
  pack_frags<<<128, 256, 0, stream>>>(Wr1, (bf16*)(frag1 + 65536),  260, 8, 8);
  pack_frags<<<128, 256, 0, stream>>>(Wl1, (bf16*)(frag1 + 131072), 260, 8, 8);
  pack_frags<<<64, 256, 0, stream>>>(Wu2, (bf16*)(frag1 + 196608), 128, 4, 8);
  pack_frags<<<64, 256, 0, stream>>>(Wr2, (bf16*)(frag1 + 229376), 128, 4, 8);
  pack_frags<<<64, 256, 0, stream>>>(Wl2, (bf16*)(frag1 + 262144), 128, 4, 8);
  pack_attrw<<<2, 256, 0, stream>>>(Wu1, (float*)(frag1 + 294912));
  pack_attrw<<<2, 256, 0, stream>>>(Wr1, (float*)(frag1 + 296960));
  pack_attrw<<<2, 256, 0, stream>>>(Wl1, (float*)(frag1 + 299008));
  pack_frags<<<64, 256, 0, stream>>>(node_w, nodeF, 128, 4, 8);
  pack_frags<<<192, 256, 0, stream>>>(W_ih, (bf16*)(frag3 + 0),      128, 4, 24);
  pack_frags<<<192, 256, 0, stream>>>(W_hh, (bf16*)(frag3 + 98304),  128, 4, 24);
  pack_frags<<<64, 256, 0, stream>>>(d1w,  (bf16*)(frag3 + 196608), 128, 4, 8);

  prep_hy<<<TLEN / 256, 256, 0, stream>>>(meas, fcw, fcb, hyB);
  dim3 gt(TLEN / 32, NF / 32), bt(32, 8);
  prep_ht<<<gt, bt, 0, stream>>>(h, hTB);

  gnn_edge_node<<<256, 256, 309248, stream>>>(hTB, hyB, ea_r, ea_u, ea_l,
      (const float*)frag1, nodeF, bu1, bu2, br1, br2, bl1, bl2, node_b, buf);

  col_stats<<<512, 128, 0, stream>>>(buf, part);
  stats_reduce<<<1, 128, 0, stream>>>(part, 512, ngam, nbet, st1);

  float* outDec = (float*)d_out;
  float* outH = outDec + (size_t)DSZ * TLEN;
  gnn_gru_dec1<<<256, 256, 263168, stream>>>(buf, hTB, h, (const float*)frag3,
      st1, b_ih, b_hh, d1b, outH);

  col_stats<<<512, 128, 0, stream>>>(buf, part);
  stats_reduce<<<1, 128, 0, stream>>>(part, 512, dgam, dbet, st2);

  gnn_dec2<<<TLEN / 8, 256, 0, stream>>>(buf, st2, d2w, d2b, outDec);
}
